// MultiHeadAttentionLayer_31945966748226
// MI455X (gfx1250) — matrix operand order repaired and checked by reading, not compiled, not run on hardware
//
#include <hip/hip_runtime.h>

// ---------------------------------------------------------------------------
// MI455X (gfx1250) multi-head attention:  out = MHA(query,key,value,mask)
//  B=2, S=2048, D=1024, H=16, dk=64.  All matmuls on v_wmma_f32_16x16x32_f16.
// ---------------------------------------------------------------------------

typedef __attribute__((ext_vector_type(16))) _Float16 v16h;
typedef __attribute__((ext_vector_type(8)))  _Float16 v8h;
typedef __attribute__((ext_vector_type(8)))  float    v8f;

union V16U { v16h v; v8h h[2]; };

#define D_MODEL 1024
#define N_HEAD  16
#define BATCH   2
#define SEQ     2048
#define DK      64

// ---- gfx1250 async global->LDS copy (ASYNCcnt), with safe fallback --------
#if __has_builtin(__builtin_amdgcn_global_load_async_to_lds_b128)
#define HAVE_ASYNC_LDS 1
#else
#define HAVE_ASYNC_LDS 0
#endif

// exact parameter type per hipcc diagnostic: int __attribute__((vector_size(16)))*
typedef int v4i_vs __attribute__((vector_size(16)));
typedef __attribute__((address_space(1))) v4i_vs* gp128_t;
typedef __attribute__((address_space(3))) v4i_vs* lp128_t;

__device__ __forceinline__ void async_copy16(const _Float16* g, _Float16* l) {
#if HAVE_ASYNC_LDS
    __builtin_amdgcn_global_load_async_to_lds_b128(
        (gp128_t)(void*)g, (lp128_t)(void*)l, 0, 0);
#else
    *(v8h*)l = *(const v8h*)g;
#endif
}

__device__ __forceinline__ void async_wait0() {
#if HAVE_ASYNC_LDS
#if __has_builtin(__builtin_amdgcn_s_wait_asynccnt)
    __builtin_amdgcn_s_wait_asynccnt(0);
#else
    asm volatile("s_wait_asynccnt 0x0" ::: "memory");
#endif
#endif
}

// ---- intra-16-lane xor butterfly via v_permlane16_b32 (VALU, no LDS) ------
__device__ __forceinline__ float xlane(float v, unsigned lo, unsigned hi, int mask) {
#if __has_builtin(__builtin_amdgcn_permlane16)
    int i = __builtin_bit_cast(int, v);
    i = __builtin_amdgcn_permlane16(i, i, lo, hi, false, false);
    return __builtin_bit_cast(float, i);
#else
    return __shfl_xor(v, mask, 32);
#endif
}

__device__ __forceinline__ float half16_max(float v) {
    v = fmaxf(v, xlane(v, 0x67452301u, 0xEFCDAB89u, 1));   // xor 1
    v = fmaxf(v, xlane(v, 0x54761032u, 0xDCFE98BAu, 2));   // xor 2
    v = fmaxf(v, xlane(v, 0x32107654u, 0xBA98FEDCu, 4));   // xor 4
    v = fmaxf(v, xlane(v, 0xFEDCBA98u, 0x76543210u, 8));   // xor 8
    return v;
}

__device__ __forceinline__ float half16_sum(float v) {
    v += xlane(v, 0x67452301u, 0xEFCDAB89u, 1);
    v += xlane(v, 0x54761032u, 0xDCFE98BAu, 2);
    v += xlane(v, 0x32107654u, 0xBA98FEDCu, 4);
    v += xlane(v, 0xFEDCBA98u, 0x76543210u, 8);
    return v;
}

// ===========================================================================
// GEMM:  Out[M][N] = A[M][K] * W[N][K]^T + bias[N]
//   A: f32 or f16 (template), W/bias: f32, Out: f16 (workspace) or f32 (d_out)
//   Block: 256 thr (8 waves, 2x4), WG tile 64x128, wave tile 32x32, K step 32.
// ===========================================================================
template<bool A_F16, bool OUT_F16>
__global__ __launch_bounds__(256, 2)
void gemm_bias_wmma(const void* __restrict__ Aptr,
                    const float* __restrict__ W,
                    const float* __restrict__ bias,
                    void* __restrict__ Out,
                    int M, int N, int K)
{
    __shared__ _Float16 Asub[64 * 40];    // [row][k] pad 32->40 (bank spread)
    __shared__ _Float16 Bsub[128 * 40];   // [n][k]   pad 32->40

    const int tid  = threadIdx.x;
    const int lane = tid & 31;
    const int wave = tid >> 5;
    const int wm   = wave >> 2;           // 0..1  (M direction)
    const int wn   = wave & 3;            // 0..3  (N direction)
    const int g    = lane >> 4;           // lane half
    const int ln   = lane & 15;
    const int m0   = blockIdx.y * 64;
    const int n0   = blockIdx.x * 128;

    v8f acc[2][2] = {};

    const int arow = tid >> 2;            // 0..63
    const int akc  = (tid & 3) * 8;       // 0,8,16,24

    for (int k0 = 0; k0 < K; k0 += 32) {
        __syncthreads();
        // ---- stage A tile 64x32 into LDS (f16) ----
        if (A_F16) {
            const _Float16* A   = (const _Float16*)Aptr;
            const _Float16* src = A + (size_t)(m0 + arow) * K + k0 + akc;
            async_copy16(src, Asub + arow * 40 + akc);    // gfx1250 async path
            if (k0 + 32 < K) __builtin_prefetch(src + 32, 0, 1);
        } else {
            const float* A   = (const float*)Aptr;
            const float* src = A + (size_t)(m0 + arow) * K + k0 + akc;
            float4 x = *(const float4*)src;
            float4 y = *(const float4*)(src + 4);
            v8h d;
            d[0] = (_Float16)x.x; d[1] = (_Float16)x.y;
            d[2] = (_Float16)x.z; d[3] = (_Float16)x.w;
            d[4] = (_Float16)y.x; d[5] = (_Float16)y.y;
            d[6] = (_Float16)y.z; d[7] = (_Float16)y.w;
            *(v8h*)(Asub + arow * 40 + akc) = d;
            if (k0 + 32 < K) __builtin_prefetch(src + 32, 0, 1);
        }
        // ---- stage B tile 128x32 (W rows are K-contiguous) ----
        #pragma unroll
        for (int i = 0; i < 2; ++i) {
            const int slot = tid + i * 256;
            const int brow = slot >> 2;          // 0..127
            const int bkc  = (slot & 3) * 8;
            const float* src = W + (size_t)(n0 + brow) * K + k0 + bkc;
            float4 x = *(const float4*)src;
            float4 y = *(const float4*)(src + 4);
            v8h d;
            d[0] = (_Float16)x.x; d[1] = (_Float16)x.y;
            d[2] = (_Float16)x.z; d[3] = (_Float16)x.w;
            d[4] = (_Float16)y.x; d[5] = (_Float16)y.y;
            d[6] = (_Float16)y.z; d[7] = (_Float16)y.w;
            *(v8h*)(Bsub + brow * 40 + bkc) = d;
            if (k0 + 32 < K) __builtin_prefetch(src + 32, 0, 1);
        }
        if (A_F16) async_wait0();
        __syncthreads();

        // ---- fragments per ISA layouts + 4 WMMAs ----
        V16U a[2], b[2];
        #pragma unroll
        for (int rr = 0; rr < 2; ++rr) {
            const _Float16* ar = Asub + (wm * 32 + rr * 16 + ln) * 40;
            a[rr].h[0] = *(const v8h*)(ar + 8 * g);        // K = 8g..8g+7
            a[rr].h[1] = *(const v8h*)(ar + 16 + 8 * g);   // K = 16+8g..
        }
        #pragma unroll
        for (int t = 0; t < 2; ++t) {
            const _Float16* br = Bsub + (wn * 32 + t * 16 + ln) * 40 + 8 * g;
            b[t].h[0] = *(const v8h*)(br);                 // K = 8g..8g+7
            b[t].h[1] = *(const v8h*)(br + 16);
        }
        #pragma unroll
        for (int rr = 0; rr < 2; ++rr)
            #pragma unroll
            for (int t = 0; t < 2; ++t)
                acc[rr][t] = __builtin_amdgcn_wmma_f32_16x16x32_f16(
                    false, a[rr].v, false, b[t].v,
                    (short)0, acc[rr][t], false, false);
    }

    // ---- epilogue: bias add, store (C layout: vgpr r -> row r+8g, col ln) ----
    #pragma unroll
    for (int rr = 0; rr < 2; ++rr) {
        #pragma unroll
        for (int t = 0; t < 2; ++t) {
            const int n  = n0 + wn * 32 + t * 16 + ln;
            const float bv = bias[n];
            #pragma unroll
            for (int r = 0; r < 8; ++r) {
                const int m = m0 + wm * 32 + rr * 16 + r + 8 * g;
                const float val = acc[rr][t][r] + bv;
                if (OUT_F16)
                    ((_Float16*)Out)[(size_t)m * N + n] = (_Float16)val;
                else
                    ((float*)Out)[(size_t)m * N + n] = val;
            }
        }
    }
}

// ===========================================================================
// Flash attention: per block = 128 query rows of one (b,h); 8 waves x 16 rows.
// Streams 32-key tiles: scores via WMMA, online softmax (permlane16 butterfly
// over the 16-lane half that holds one row), P relayout via per-wave LDS,
// P*V WMMA.  K tile staged with async global->LDS.
// ===========================================================================
__global__ __launch_bounds__(256, 2)
void flash_attn_wmma(const _Float16* __restrict__ Q,
                     const _Float16* __restrict__ Km,
                     const _Float16* __restrict__ Vm,
                     const int* __restrict__ mask,
                     _Float16* __restrict__ Ctx)
{
    __shared__ _Float16 Kt[32 * 72];        // [key][d]   32x64, pad->72
    __shared__ _Float16 Vt[64 * 40];        // [d][key]   64x32, pad->40
    __shared__ _Float16 Pb[8 * 16 * 40];    // per-wave P staging 16x32 pad->40

    const int tid  = threadIdx.x;
    const int lane = tid & 31;
    const int wave = tid >> 5;
    const int g    = lane >> 4;
    const int ln   = lane & 15;

    const int qt = blockIdx.x;              // 0..15  (128 rows each)
    const int bh = blockIdx.y;              // 0..31
    const int b  = bh >> 4;
    const int h  = bh & 15;

    const size_t rowbase = (size_t)b * SEQ;
    const int    qrow    = qt * 128 + wave * 16 + ln;

    // Q fragment: 16x64 = two 16x32 A-operand chunks, resident in VGPRs
    V16U qa[2];
    {
        const _Float16* qp = Q + (rowbase + qrow) * D_MODEL + h * DK;
        #pragma unroll
        for (int c = 0; c < 2; ++c) {
            qa[c].h[0] = *(const v8h*)(qp + c * 32 + 8 * g);
            qa[c].h[1] = *(const v8h*)(qp + c * 32 + 16 + 8 * g);
        }
    }

    v8f   o[4] = {};                        // 16x64 output accumulator
    float m_i[8], l_i[8];
    #pragma unroll
    for (int r = 0; r < 8; ++r) { m_i[r] = -1.0e30f; l_i[r] = 0.0f; }

    _Float16* Pw = Pb + wave * 16 * 40;

    const int skey = tid >> 3;              // 0..31
    const int sdc  = (tid & 7) * 8;         // 0..56

    for (int j0 = 0; j0 < SEQ; j0 += 32) {
        __syncthreads();
        // ---- stage K [32 keys][64] (async) and V^T [64][32 keys] in LDS ----
        {
            const _Float16* kp = Km + (rowbase + j0 + skey) * D_MODEL + h * DK + sdc;
            async_copy16(kp, Kt + skey * 72 + sdc);
            const _Float16* vp = Vm + (rowbase + j0 + skey) * D_MODEL + h * DK + sdc;
            v8h vr = *(const v8h*)vp;
            #pragma unroll
            for (int i = 0; i < 8; ++i) Vt[(sdc + i) * 40 + skey] = vr[i];
        }
        async_wait0();
        __syncthreads();

        // ---- scores S = Q K^T : 16x32 (two 16x16 C tiles), 4 WMMAs ----
        v8f sc[2] = {};
        #pragma unroll
        for (int c = 0; c < 2; ++c) {
            #pragma unroll
            for (int t = 0; t < 2; ++t) {
                V16U kb;   // B operand: col = key (t*16+ln), K rows contiguous
                const _Float16* kr = Kt + (t * 16 + ln) * 72 + c * 32 + 8 * g;
                kb.h[0] = *(const v8h*)(kr);
                kb.h[1] = *(const v8h*)(kr + 16);
                sc[t] = __builtin_amdgcn_wmma_f32_16x16x32_f16(
                    false, qa[c].v, false, kb.v, (short)0, sc[t], false, false);
            }
        }

        // ---- scale, mask, online softmax (row = r + 8g, cols across half) --
        const int mv0 = mask[rowbase + j0 + ln];
        const int mv1 = mask[rowbase + j0 + 16 + ln];
        #pragma unroll
        for (int r = 0; r < 8; ++r) {
            float s0 = sc[0][r] * 0.125f;               // 1/sqrt(64)
            float s1 = sc[1][r] * 0.125f;
            s0 = (mv0 == 0) ? -10000.0f : s0;
            s1 = (mv1 == 0) ? -10000.0f : s1;
            const float smax = half16_max(fmaxf(s0, s1));
            const float mn   = fmaxf(m_i[r], smax);
            const float p0   = __expf(s0 - mn);
            const float p1   = __expf(s1 - mn);
            const float ps   = half16_sum(p0 + p1);
            const float corr = __expf(m_i[r] - mn);
            l_i[r] = l_i[r] * corr + ps;
            m_i[r] = mn;
            #pragma unroll
            for (int t4 = 0; t4 < 4; ++t4) o[t4][r] *= corr;
            // park P (C layout -> LDS row-major) for A-operand reload
            const int pm = r + 8 * g;
            Pw[pm * 40 + ln]      = (_Float16)p0;
            Pw[pm * 40 + 16 + ln] = (_Float16)p1;
        }
        __builtin_amdgcn_wave_barrier();   // LDS is in-order per wave (DScnt)

        // ---- O += P V : P as 16x32 A frag, V^T as four 32x16 B frags ----
        V16U pa;
        const _Float16* pr = Pw + ln * 40;
        pa.h[0] = *(const v8h*)(pr + 8 * g);
        pa.h[1] = *(const v8h*)(pr + 16 + 8 * g);
        #pragma unroll
        for (int t = 0; t < 4; ++t) {
            V16U vb;
            const _Float16* vr2 = Vt + (t * 16 + ln) * 40 + 8 * g;
            vb.h[0] = *(const v8h*)(vr2);
            vb.h[1] = *(const v8h*)(vr2 + 16);
            o[t] = __builtin_amdgcn_wmma_f32_16x16x32_f16(
                false, pa.v, false, vb.v, (short)0, o[t], false, false);
        }
        __builtin_amdgcn_wave_barrier();
    }

    // ---- normalize and store context in [b,s,h*dk] layout (f16 ws) ----
    #pragma unroll
    for (int t = 0; t < 4; ++t) {
        #pragma unroll
        for (int r = 0; r < 8; ++r) {
            const int m    = r + 8 * g;
            const int srow = qt * 128 + wave * 16 + m;
            const float val = o[t][r] / l_i[r];
            Ctx[(rowbase + srow) * D_MODEL + h * DK + t * 16 + ln] = (_Float16)val;
        }
    }
}

// ===========================================================================
// Host launcher
// ===========================================================================
extern "C" void kernel_launch(void* const* d_in, const int* in_sizes, int n_in,
                              void* d_out, int out_size, void* d_ws, size_t ws_size,
                              hipStream_t stream)
{
    (void)in_sizes; (void)n_in; (void)out_size; (void)ws_size;

    const float* query = (const float*)d_in[0];
    const float* key   = (const float*)d_in[1];
    const float* value = (const float*)d_in[2];
    const int*   mask  = (const int*)  d_in[3];
    const float* Wq = (const float*)d_in[4];  const float* bq = (const float*)d_in[5];
    const float* Wk = (const float*)d_in[6];  const float* bk = (const float*)d_in[7];
    const float* Wv = (const float*)d_in[8];  const float* bv = (const float*)d_in[9];
    const float* Wo = (const float*)d_in[10]; const float* bo = (const float*)d_in[11];
    float* out = (float*)d_out;

    const size_t ACT = (size_t)BATCH * SEQ * D_MODEL;   // 4096*1024
    _Float16* Q16 = (_Float16*)d_ws;
    _Float16* K16 = Q16 + ACT;
    _Float16* V16 = K16 + ACT;
    _Float16* C16 = V16 + ACT;                          // 32 MB total f16 ws

    const int M = BATCH * SEQ;                          // 4096
    const dim3 gemm_grid(D_MODEL / 128, M / 64);        // (8, 64)
    const dim3 attn_grid(SEQ / 128, BATCH * N_HEAD);    // (16, 32)

    // 1) QKV projections (f32 in -> f16 ws)
    gemm_bias_wmma<false, true><<<gemm_grid, 256, 0, stream>>>(
        query, Wq, bq, Q16, M, D_MODEL, D_MODEL);
    gemm_bias_wmma<false, true><<<gemm_grid, 256, 0, stream>>>(
        key, Wk, bk, K16, M, D_MODEL, D_MODEL);
    gemm_bias_wmma<false, true><<<gemm_grid, 256, 0, stream>>>(
        value, Wv, bv, V16, M, D_MODEL, D_MODEL);

    // 2) Flash attention (writes context in [b,s,h*dk], i.e. merged-head layout)
    flash_attn_wmma<<<attn_grid, 256, 0, stream>>>(Q16, K16, V16, mask, C16);

    // 3) Output projection (f16 ws -> f32 d_out)
    gemm_bias_wmma<true, false><<<gemm_grid, 256, 0, stream>>>(
        C16, Wo, bo, out, M, D_MODEL, D_MODEL);
}
